// S4R_58308476010695
// MI455X (gfx1250) — compile-verified, matrix operand order checked by reading, not run on hardware
//
#include <hip/hip_runtime.h>

#define B_ 8
#define H_ 256
#define L_ 4096
#define N_ 64
#define NC_ 128   // chunks per sequence (L/32)
#define LDSK 40   // padded LDS row stride (bf16) for mix kernel B tiles

typedef __attribute__((ext_vector_type(16))) __bf16 v16bf;
typedef __attribute__((ext_vector_type(8)))  __bf16 v8bf;
typedef __attribute__((ext_vector_type(2)))  __bf16 v2bf;
typedef __attribute__((ext_vector_type(8)))  float  v8f;

// native f32->bf16 convert; pairs should form v_cvt_pk_bf16_f32
static __device__ __forceinline__ unsigned int pack_bf16(float a, float b) {
  v2bf v;
  v[0] = (__bf16)a;
  v[1] = (__bf16)b;
  return __builtin_bit_cast(unsigned int, v);
}
static __device__ __forceinline__ unsigned short f32_to_bf16(float f) {
  return __builtin_bit_cast(unsigned short, (__bf16)f);
}
static __device__ __forceinline__ float bf16_to_f32(unsigned short s) {
  return __builtin_bit_cast(float, (unsigned int)s << 16);
}
static __device__ __forceinline__ v16bf mk_frag(const v8bf lo, const v8bf hi) {
  return __builtin_shufflevector(lo, hi, 0, 1, 2, 3, 4, 5, 6, 7,
                                 8, 9, 10, 11, 12, 13, 14, 15);
}

// ---------------------------------------------------------------------------
// Kernel P: per-head chunked-scan operators (bf16, row-major = WMMA-A friendly)
//  P[h][row 0..127][k 0..31] : Lam^(31-k)    (rows 0-63 re, 64-127 im)
//  G[h][j 0..31][col 0..127] : [2Re(CB*Lam^(j+1)) | -2Im(CB*Lam^(j+1))]
//  T[h][j][k] = K[h][j-k] (k<=j) else 0, K[d] = 2Re(sum_n CB*Lam^d)
// ---------------------------------------------------------------------------
__global__ __launch_bounds__(64) void s4r_precompute(
    const float* __restrict__ lam_re, const float* __restrict__ lam_im,
    const float* __restrict__ cb_re,  const float* __restrict__ cb_im,
    unsigned short* __restrict__ Tbuf,   // [H][32][32]
    unsigned short* __restrict__ Pbuf,   // [H][128][32]
    unsigned short* __restrict__ Gbuf)   // [H][32][128]
{
  const int h = blockIdx.x;
  const int n = threadIdx.x;             // 0..63 (one state per thread)
  __shared__ float contrib[64][33];      // padded: conflict-free transpose
  __shared__ float Kk[32];

  const float lr = lam_re[h * N_ + n], li = lam_im[h * N_ + n];
  const float cr = cb_re[h * N_ + n],  ci = cb_im[h * N_ + n];

  unsigned short* Pr = Pbuf + ((size_t)h * 128 + n) * 32;        // re row
  unsigned short* Pi = Pbuf + ((size_t)h * 128 + 64 + n) * 32;   // im row

  float pr = 1.f, pi = 0.f;              // Lam^0
  for (int j = 0; j < 32; ++j) {
    Pr[31 - j] = f32_to_bf16(pr);        // P[n][k] = Lam^(31-k)
    Pi[31 - j] = f32_to_bf16(pi);
    contrib[n][j] = 2.f * (cr * pr - ci * pi);   // -> K[j]
    float nr = pr * lr - pi * li;        // advance to Lam^(j+1)
    float ni = pr * li + pi * lr;
    pr = nr; pi = ni;
    Gbuf[((size_t)h * 32 + j) * 128 + n]      = f32_to_bf16( 2.f * (cr * pr - ci * pi));
    Gbuf[((size_t)h * 32 + j) * 128 + 64 + n] = f32_to_bf16(-2.f * (cr * pi + ci * pr));
  }
  __syncthreads();
  if (n < 32) {                          // deterministic transpose-reduce
    float s = 0.f;
    for (int m = 0; m < 64; ++m) s += contrib[m][n];
    Kk[n] = s;
  }
  __syncthreads();
  for (int e = n; e < 1024; e += 64) {   // T lower-triangular Toeplitz
    const int j = e >> 5, k = e & 31;
    Tbuf[((size_t)h * 32 + j) * 32 + k] = f32_to_bf16((k <= j) ? Kk[j - k] : 0.f);
  }
}

// ---------------------------------------------------------------------------
// Kernel W: one-shot pack of W_mix (f32 [512,256]) to bf16 row-major.
// ---------------------------------------------------------------------------
__global__ __launch_bounds__(256) void s4r_pack_w(
    const float* __restrict__ W, unsigned short* __restrict__ Wbf) {
  const int i = blockIdx.x * 256 + threadIdx.x;
  Wbf[i] = f32_to_bf16(W[i]);
}

// ---------------------------------------------------------------------------
// Kernel S: chunked SSM scan on WMMA. One block per (b,h):
//   op2: S[c] = P*U[c]           (64 WMMAs, K=32)
//   prefix: sin[c] = Lam^32 (.) sin[c-1] + sloc[c-1]  (127 serial steps only)
//   op1+op4: Y = T*U + G*Sin; y = tanh(Y + D*u) -> bf16   (80 WMMAs)
// Slot map in sbuf[129][128]: slot0 = 0; op2 writes sloc[c] -> slot c+1;
// prefix overwrites slot c with sin[c]; op4 chunk c reads slot c.
// ---------------------------------------------------------------------------
__global__ __launch_bounds__(256) void s4r_ssm_wmma(
    const float* __restrict__ u,
    const float* __restrict__ lam_re, const float* __restrict__ lam_im,
    const float* __restrict__ Dv,
    const unsigned short* __restrict__ Tbuf,
    const unsigned short* __restrict__ Pbuf,
    const unsigned short* __restrict__ Gbuf,
    unsigned short* __restrict__ y_bf16)
{
  const int bh   = blockIdx.x;
  const int h    = bh & (H_ - 1);
  const int lane = threadIdx.x & 31;
  const int wv   = threadIdx.x >> 5;
  const int nIdx  = lane & 15;
  const int hiSel = lane >> 4;
  const int kb    = hiSel * 8;

  __shared__ unsigned short ubuf[L_];          // u row in bf16, 8 KB
  __shared__ unsigned short sbuf[129 * 128];   // state slots, 33 KB

  const float* urow = u + (size_t)bh * L_;

  // stage u -> bf16 LDS (b128 loads, b64 stores); zero slot 0
  #pragma unroll
  for (int i = 0; i < 4; ++i) {
    const int idx = (threadIdx.x + i * 256) * 4;
    const float4 v = *(const float4*)(urow + idx);
    uint2 d;
    d.x = pack_bf16(v.x, v.y);
    d.y = pack_bf16(v.z, v.w);
    *(uint2*)&ubuf[idx] = d;
  }
  if (threadIdx.x < 128) sbuf[threadIdx.x] = 0;
  __syncthreads();

  // ---- op2: S = P*U ; wave wv owns m-tile wv (16 of 128 state rows) ----
  {
    const unsigned short* Ph = Pbuf + (size_t)h * 128 * 32;
    const int rowm = wv * 16 + nIdx;
    const v16bf afrag = mk_frag(*(const v8bf*)(Ph + rowm * 32 + kb),
                                *(const v8bf*)(Ph + rowm * 32 + 16 + kb));
    const int mbase = wv * 16 + hiSel * 8;
    #pragma unroll
    for (int nt = 0; nt < 8; ++nt) {
      const int c = nt * 16 + nIdx;
      const v16bf bfrag = mk_frag(*(const v8bf*)&ubuf[c * 32 + kb],
                                  *(const v8bf*)&ubuf[c * 32 + 16 + kb]);
      v8f acc = {};
      acc = __builtin_amdgcn_wmma_f32_16x16x32_bf16(
          false, afrag, false, bfrag, (short)0, acc, false, false);
      uint4 pk;
      pk.x = pack_bf16(acc[0], acc[1]);
      pk.y = pack_bf16(acc[2], acc[3]);
      pk.z = pack_bf16(acc[4], acc[5]);
      pk.w = pack_bf16(acc[6], acc[7]);
      *(uint4*)&sbuf[(c + 1) * 128 + mbase] = pk;   // sloc[c] -> slot c+1
    }
  }
  __syncthreads();

  // ---- prefix over chunks: only 127 serial steps, 64 lanes active ----
  if (threadIdx.x < 64) {
    const int n = threadIdx.x;
    float qr = lam_re[h * N_ + n], qi = lam_im[h * N_ + n];
    #pragma unroll
    for (int s5 = 0; s5 < 5; ++s5) {      // Lam^32 by repeated squaring
      const float nr = qr * qr - qi * qi;
      qi = 2.f * qr * qi; qr = nr;
    }
    float sr = 0.f, si = 0.f;             // sin[0] = 0
    for (int c = 1; c < NC_; ++c) {
      const float ar = bf16_to_f32(sbuf[c * 128 + n]);        // sloc[c-1]
      const float ai = bf16_to_f32(sbuf[c * 128 + 64 + n]);
      const float nr = qr * sr - qi * si + ar;
      const float ni = qr * si + qi * sr + ai;
      sr = nr; si = ni;
      sbuf[c * 128 + n]      = f32_to_bf16(sr);               // sin[c]
      sbuf[c * 128 + 64 + n] = f32_to_bf16(si);
    }
  }
  __syncthreads();

  // ---- op1+op4: Y = T*U + G*Sin ; +D*u, tanh ; wave wv owns chunk tile wv ----
  {
    const float dh = Dv[h];
    const unsigned short* Th = Tbuf + (size_t)h * 32 * 32;
    const unsigned short* Gh = Gbuf + (size_t)h * 32 * 128;
    unsigned short* yb = y_bf16 + (size_t)bh * L_;
    const int c = wv * 16 + nIdx;          // lane's chunk column
    #pragma unroll
    for (int mt = 0; mt < 2; ++mt) {
      const int rowm = mt * 16 + nIdx;
      const v16bf at = mk_frag(*(const v8bf*)(Th + rowm * 32 + kb),
                               *(const v8bf*)(Th + rowm * 32 + 16 + kb));
      const v16bf bu = mk_frag(*(const v8bf*)&ubuf[c * 32 + kb],
                               *(const v8bf*)&ubuf[c * 32 + 16 + kb]);
      v8f acc = {};
      acc = __builtin_amdgcn_wmma_f32_16x16x32_bf16(
          false, at, false, bu, (short)0, acc, false, false);
      #pragma unroll
      for (int kk = 0; kk < 4; ++kk) {
        const v16bf ag = mk_frag(
            *(const v8bf*)(Gh + rowm * 128 + kk * 32 + kb),
            *(const v8bf*)(Gh + rowm * 128 + kk * 32 + 16 + kb));
        const v16bf bs = mk_frag(
            *(const v8bf*)&sbuf[c * 128 + kk * 32 + kb],
            *(const v8bf*)&sbuf[c * 128 + kk * 32 + 16 + kb]);
        acc = __builtin_amdgcn_wmma_f32_16x16x32_bf16(
            false, ag, false, bs, (short)0, acc, false, false);
      }
      // epilogue: lane's 8 rows are 8 consecutive l -> one b128 store
      const int lb = c * 32 + mt * 16 + hiSel * 8;
      float yv[8];
      #pragma unroll
      for (int r = 0; r < 8; ++r)
        yv[r] = tanhf(acc[r] + dh * bf16_to_f32(ubuf[lb + r]));
      uint4 pk;
      pk.x = pack_bf16(yv[0], yv[1]);
      pk.y = pack_bf16(yv[2], yv[3]);
      pk.z = pack_bf16(yv[4], yv[5]);
      pk.w = pack_bf16(yv[6], yv[7]);
      *(uint4*)&yb[lb] = pk;
    }
  }
}

// ---------------------------------------------------------------------------
// Kernel M: 1x1 mixing matmul (WMMA bf16) + fused GLU.
// ---------------------------------------------------------------------------
__global__ __launch_bounds__(256) void s4r_mix_glu(
    const unsigned short* __restrict__ y,    // [B,H,L] bf16
    const unsigned short* __restrict__ Wbf,  // [512,256] bf16
    const float* __restrict__ bmix,
    float* __restrict__ out)
{
  const int l0   = blockIdx.x * 16;
  const int b    = blockIdx.y;
  const int mb   = blockIdx.z;
  const int wv   = threadIdx.x >> 5;
  const int lane = threadIdx.x & 31;
  const int m0   = mb * 128 + wv * 16;
  const int nIdx  = lane & 15;
  const int hiSel = lane >> 4;
  const int kb    = hiSel * 8;

  __shared__ unsigned short lds_b[16 * LDSK];

  v8f acc_a = {}; v8f acc_g = {};
  const int row_a = m0 + nIdx;
  const unsigned short* wa = Wbf + (size_t)row_a * H_;
  const unsigned short* wg = Wbf + (size_t)(row_a + 256) * H_;

  const int sk  = threadIdx.x >> 3;
  const int sn2 = (threadIdx.x & 7) * 2;
  const unsigned short* ysrc = y + ((size_t)(b * H_ + sk)) * L_ + l0 + sn2;

  for (int kk = 0; kk < 8; ++kk) {
    const int k0 = kk * 32;
    {
      const unsigned int v = *(const unsigned int*)(ysrc + (size_t)k0 * L_);
      lds_b[sn2 * LDSK + sk]       = (unsigned short)(v & 0xFFFFu);
      lds_b[(sn2 + 1) * LDSK + sk] = (unsigned short)(v >> 16);
    }
    __syncthreads();
    if (kk < 7) __builtin_prefetch(wa + k0 + 32, 0, 1);

    const v16bf bfrag = mk_frag(*(const v8bf*)&lds_b[nIdx * LDSK + kb],
                                *(const v8bf*)&lds_b[nIdx * LDSK + 16 + kb]);
    const v16bf afrag_a = mk_frag(*(const v8bf*)(wa + k0 + kb),
                                  *(const v8bf*)(wa + k0 + 16 + kb));
    const v16bf afrag_g = mk_frag(*(const v8bf*)(wg + k0 + kb),
                                  *(const v8bf*)(wg + k0 + 16 + kb));
    acc_a = __builtin_amdgcn_wmma_f32_16x16x32_bf16(
        false, afrag_a, false, bfrag, (short)0, acc_a, false, false);
    acc_g = __builtin_amdgcn_wmma_f32_16x16x32_bf16(
        false, afrag_g, false, bfrag, (short)0, acc_g, false, false);
    __syncthreads();
  }

  #pragma unroll
  for (int r = 0; r < 8; ++r) {
    const int hch = m0 + r + hiSel * 8;
    const float za = acc_a[r] + bmix[hch];
    const float zg = acc_g[r] + bmix[hch + 256];
    const float sg = 1.f / (1.f + __expf(-zg));
    out[((size_t)(b * H_ + hch)) * L_ + l0 + nIdx] = za * sg;
  }
}

// ---------------------------------------------------------------------------
extern "C" void kernel_launch(void* const* d_in, const int* in_sizes, int n_in,
                              void* d_out, int out_size, void* d_ws, size_t ws_size,
                              hipStream_t stream) {
  (void)in_sizes; (void)n_in; (void)out_size; (void)ws_size;
  const float* u   = (const float*)d_in[0];
  const float* lre = (const float*)d_in[1];
  const float* lim = (const float*)d_in[2];
  const float* cre = (const float*)d_in[3];
  const float* cim = (const float*)d_in[4];
  const float* Dv  = (const float*)d_in[5];
  const float* W   = (const float*)d_in[6];
  const float* bm  = (const float*)d_in[7];
  float* out = (float*)d_out;

  char* ws = (char*)d_ws;
  unsigned short* y_ws = (unsigned short*)ws;                       ws += (size_t)B_ * H_ * L_ * 2;
  unsigned short* w_bf = (unsigned short*)ws;                       ws += (size_t)512 * H_ * 2;
  unsigned short* Tbuf = (unsigned short*)ws;                       ws += (size_t)H_ * 32 * 32 * 2;
  unsigned short* Pbuf = (unsigned short*)ws;                       ws += (size_t)H_ * 128 * 32 * 2;
  unsigned short* Gbuf = (unsigned short*)ws;

  s4r_precompute<<<dim3(H_), 64, 0, stream>>>(lre, lim, cre, cim, Tbuf, Pbuf, Gbuf);
  s4r_pack_w<<<dim3((512 * H_) / 256), 256, 0, stream>>>(W, w_bf);
  s4r_ssm_wmma<<<dim3(B_ * H_), 256, 0, stream>>>(u, lre, lim, Dv,
                                                  Tbuf, Pbuf, Gbuf, y_ws);
  dim3 grid(L_ / 16, B_, 2);
  s4r_mix_glu<<<grid, 256, 0, stream>>>(y_ws, w_bf, bm, out);
}
